// SMGStack_65738769433057
// MI455X (gfx1250) — matrix operand order repaired and checked by reading, not compiled, not run on hardware
//
#include <hip/hip_runtime.h>
#include <math.h>

typedef _Float16 half_t;
typedef __attribute__((ext_vector_type(16))) _Float16 v16h;
typedef __attribute__((ext_vector_type(8)))  float    v8f;
typedef __attribute__((ext_vector_type(4)))  float    v4f;

#define NNODES 100000
#define NEDGES 640000
#define DFEAT  128
#define NLAYER 3

// ---------------- utility / elementwise kernels ----------------

// Convert + swizzle a [K,128] f32 weight matrix into WMMA-fragment order f16:
//   out[(((nt*KT + kt)*32 + lane)*16) + h] = (f16) in[k*128 + n]
//   with k = kt*32 + ((lane>=16)?16:0) + h ,  n = nt*16 + (lane&15)
// so each lane's 16-half B fragment is one contiguous, 32B-aligned block.
__global__ void k_swz_w(const float* __restrict__ in, half_t* __restrict__ out,
                        int KT, int total /* = K*128 */) {
  int i = blockIdx.x * blockDim.x + threadIdx.x;
  if (i >= total) return;
  int h    = i & 15;
  int lane = (i >> 4) & 31;
  int t    = i >> 9;
  int kt   = t % KT;
  int nt   = t / KT;
  int k = kt * 32 + ((lane >> 4) << 4) + h;
  int n = nt * 16 + (lane & 15);
  out[i] = (half_t)in[(size_t)k * 128 + n];
}

__global__ void k_init(const float* __restrict__ xin, float* __restrict__ X,
                       float* __restrict__ M, int n) {
  int i = blockIdx.x * blockDim.x + threadIdx.x;
  if (i < n) { X[i] = xin[i]; M[i] = 1.0f; }
}

__global__ void k_mul(const float* __restrict__ a, const float* __restrict__ b,
                      float* __restrict__ o, int n) {
  int i = blockIdx.x * blockDim.x + threadIdx.x;
  if (i < n) o[i] = a[i] * b[i];
}

// zero a 128-wide column block inside a row-major buffer with leading dim ld
__global__ void k_zero_cols(float* __restrict__ p, int ld, int n /* rows*128 */) {
  int i = blockIdx.x * blockDim.x + threadIdx.x;
  if (i < n) {
    int r = i >> 7, c = i & 127;
    p[(long)r * ld + c] = 0.0f;
  }
}

// X = relu((agg + lin) * M) ; agg has leading dim 256
__global__ void k_combine(const float* __restrict__ aggp, const float* __restrict__ lin,
                          const float* __restrict__ M, float* __restrict__ X, int n) {
  int i = blockIdx.x * blockDim.x + threadIdx.x;
  if (i < n) {
    int r = i >> 7, c = i & 127;
    float v = (aggp[(long)r * 256 + c] + lin[i]) * M[i];
    X[i] = fmaxf(v, 0.0f);
  }
}

// out[dst[e],:] += edge_attr[e] * h[src[e],:]   (one wave per edge, float4 per lane)
__global__ void k_agg(const float* __restrict__ h, int ldh,
                      const int* __restrict__ srcI, const int* __restrict__ dstI,
                      const float* __restrict__ ea, float* __restrict__ out, int ldo) {
  long gid = (long)blockIdx.x * blockDim.x + threadIdx.x;
  int e = (int)(gid >> 5);
  if (e >= NEDGES) return;
  int c4 = ((int)gid & 31) << 2;            // feature offset, 4 floats per lane
  int s = srcI[e], d = dstI[e];
  float w = ea[e];
  v4f v = *(const v4f*)(h + (long)s * ldh + c4);
  float* op = out + (long)d * ldo + c4;
  atomicAdd(op + 0, w * v[0]);
  atomicAdd(op + 1, w * v[1]);
  atomicAdd(op + 2, w * v[2]);
  atomicAdd(op + 3, w * v[3]);
}

// ---------------- WMMA GEMM: D[nrows,128] = act(A[nrows,K] @ W[K,128] + bias) ----------------
// KT = K/32 ; ACT: 0=none 1=relu 2=sigmoid ; PRERELU: relu applied to A on load
// W is in fragment-swizzled f16 order (see k_swz_w) and is staged in LDS once
// per workgroup (all 8 waves share it); B fragments come from ds_load_b128.
// One wave computes a 16x128 output slab with 8 independent accumulators
// (kt-outer / nt-inner -> consecutive WMMAs are independent, no hazard NOPs).
// A fragments fully preloaded -> in-place GEMM (Ap == Dp) is safe.
template <int KT, int ACT, bool PRERELU, bool HASBIAS>
__global__ void gemm_wmma(const float* __restrict__ Ap, int lda,
                          const half_t* __restrict__ Wp,   // swizzled [KT*32,128] f16
                          const float* __restrict__ bias,  // [128] or null
                          float* __restrict__ Dp, int ldd, int nrows) {
  extern __shared__ half_t lds_w[];                // KT*4096 halves

  int lane    = threadIdx.x & 31;
  int wave    = threadIdx.x >> 5;
  int rowTile = blockIdx.x * (blockDim.x >> 5) + wave;
  int mBase   = rowTile * 16;
  bool active = (mBase < nrows);                   // wave-uniform

  // --- cooperative stage of swizzled weights into LDS (32B chunks) ---
  {
    const v16h* gsrc = (const v16h*)Wp;
    v16h* ldst = (v16h*)lds_w;
#pragma unroll
    for (int c = 0; c < KT; ++c)                   // KT*256 chunks / 256 threads
      ldst[c * 256 + threadIdx.x] = gsrc[c * 256 + threadIdx.x];
  }

  int halfSel = lane >> 4;                         // 0 or 1
  int lm      = lane & 15;

  // --- preload + convert A fragments (overlaps the LDS staging) ---
  v16h afrag[KT];
  if (active) {
    const float* arow = Ap + (long)(mBase + lm) * lda;
#pragma unroll
    for (int kt = 0; kt < KT; ++kt) {
      int k0 = kt * 32 + (halfSel << 4);           // halves 0..15 -> K=k0..k0+15
      v4f x0 = *(const v4f*)(arow + k0);
      v4f x1 = *(const v4f*)(arow + k0 + 4);
      v4f x2 = *(const v4f*)(arow + k0 + 8);
      v4f x3 = *(const v4f*)(arow + k0 + 12);
      if (PRERELU) {
#pragma unroll
        for (int j = 0; j < 4; ++j) {
          x0[j] = fmaxf(x0[j], 0.0f);
          x1[j] = fmaxf(x1[j], 0.0f);
          x2[j] = fmaxf(x2[j], 0.0f);
          x3[j] = fmaxf(x3[j], 0.0f);
        }
      }
      v16h a;
#pragma unroll
      for (int j = 0; j < 4; ++j) {
        a[j]      = (half_t)x0[j];
        a[4 + j]  = (half_t)x1[j];
        a[8 + j]  = (half_t)x2[j];
        a[12 + j] = (half_t)x3[j];
      }
      afrag[kt] = a;
    }
  }

  __syncthreads();
  if (!active) return;                             // wave-uniform: EXEC stays all-ones

  // --- 8 independent accumulators, kt-outer / nt-inner ---
  v8f acc[8];
#pragma unroll
  for (int nt = 0; nt < 8; ++nt) acc[nt] = (v8f){};

  const half_t* wlane = lds_w + (size_t)lane * 16; // lane's fragment slot in LDS
#pragma unroll
  for (int kt = 0; kt < KT; ++kt) {
#pragma unroll
    for (int nt = 0; nt < 8; ++nt) {
      // one contiguous 32B LDS read per fragment (ds_load_b128 x2)
      v16h b = *(const v16h*)(wlane + (size_t)(nt * KT + kt) * 32 * 16);
      acc[nt] = __builtin_amdgcn_wmma_f32_16x16x32_f16(
          false, afrag[kt], false, b, (short)0, acc[nt], false, false);
    }
  }

  // --- epilogue ---
#pragma unroll
  for (int nt = 0; nt < 8; ++nt) {
    int nBase = nt * 16;
    float bv = 0.0f;
    if (HASBIAS) bv = bias[nBase + lm];
#pragma unroll
    for (int r = 0; r < 8; ++r) {
      float v = acc[nt][r] + bv;
      if (ACT == 1) v = fmaxf(v, 0.0f);
      else if (ACT == 2) v = 1.0f / (1.0f + __expf(-v));
      int row = mBase + r + (halfSel << 3);
      Dp[(long)row * ldd + nBase + lm] = v;
    }
  }
}

// ---------------- launcher ----------------

extern "C" void kernel_launch(void* const* d_in, const int* in_sizes, int n_in,
                              void* d_out, int out_size, void* d_ws, size_t ws_size,
                              hipStream_t stream) {
  (void)in_sizes; (void)n_in; (void)out_size; (void)ws_size;

  const float* x     = (const float*)d_in[0];
  const float* eattr = (const float*)d_in[1];
  const int*   eidx  = (const int*)d_in[2];
  const float* convw = (const float*)d_in[3];
  const float* convl = (const float*)d_in[4];
  const float* l1w   = (const float*)d_in[5];
  const float* l1b   = (const float*)d_in[6];
  const float* l2w   = (const float*)d_in[7];
  const float* l2b   = (const float*)d_in[8];
  const float* m1w   = (const float*)d_in[9];
  const float* m1b   = (const float*)d_in[10];
  const float* m2w   = (const float*)d_in[11];
  const float* m2b   = (const float*)d_in[12];
  const float* p1w   = (const float*)d_in[13];
  const float* p1b   = (const float*)d_in[14];
  const float* p2w   = (const float*)d_in[15];
  const float* p2b   = (const float*)d_in[16];
  const int* srcI = eidx;
  const int* dstI = eidx + NEDGES;

  // workspace layout: [swizzled f16 weights ~0.75MB | X | M | A | Wcat(Nx256)]
  char* ws = (char*)d_ws;
  half_t* wf16 = (half_t*)ws;
  half_t* f_l1 = wf16 + 0;        // 3*16384
  half_t* f_l2 = wf16 + 49152;    // 3*16384
  half_t* f_m1 = wf16 + 98304;    // 3*32768
  half_t* f_m2 = wf16 + 196608;   // 3*16384
  half_t* f_cw = wf16 + 245760;   // 3*16384
  half_t* f_cl = wf16 + 294912;   // 3*16384
  half_t* f_p1 = wf16 + 344064;   // 16384
  half_t* f_p2 = wf16 + 360448;   // 16384

  const size_t unit = (size_t)NNODES * DFEAT * sizeof(float);
  float* X  = (float*)(ws + (1 << 20));
  float* M  = (float*)(ws + (1 << 20) + unit);
  float* A  = (float*)(ws + (1 << 20) + 2 * unit);
  float* WC = (float*)(ws + (1 << 20) + 3 * unit);   // N x 256

  const int ND  = NNODES * DFEAT;                 // 12.8M
  const int EWB = (ND + 255) / 256;
  const int GG  = ((NNODES / 16) + 7) / 8;        // 782 blocks, 8 waves/block
  const int AGB = (NEDGES * 32 + 255) / 256;
  const int S4  = 4 * 8192;                       // LDS bytes, KT=4 (32KB)
  const int S8  = 8 * 8192;                       // LDS bytes, KT=8 (64KB)

  // --- convert + swizzle all weights to f16 fragment order (per 128-col matrix) ---
#define SWZ(src, dst, KT, elems) \
  k_swz_w<<<((elems) + 255) / 256, 256, 0, stream>>>((src), (dst), (KT), (elems))
  for (int i = 0; i < NLAYER; ++i) {
    SWZ(l1w   + i * 16384, f_l1 + i * 16384, 4, 16384);
    SWZ(l2w   + i * 16384, f_l2 + i * 16384, 4, 16384);
    SWZ(m1w   + i * 32768, f_m1 + i * 32768, 8, 32768);
    SWZ(m2w   + i * 16384, f_m2 + i * 16384, 4, 16384);
    SWZ(convw + i * 16384, f_cw + i * 16384, 4, 16384);
    SWZ(convl + i * 16384, f_cl + i * 16384, 4, 16384);
  }
  SWZ(p1w, f_p1, 4, 16384);
  SWZ(p2w, f_p2, 4, 16384);
#undef SWZ

  k_init<<<EWB, 256, 0, stream>>>(x, X, M, ND);

  for (int i = 0; i < NLAYER; ++i) {
    // xm = x * mask
    k_mul<<<EWB, 256, 0, stream>>>(X, M, A, ND);
    // h = xm @ l1 + b1   (store in M's buffer; old mask no longer needed)
    gemm_wmma<4, 0, false, true><<<GG, 256, S4, stream>>>(
        A, 128, f_l1 + i * 16384, l1b + i * 128, M, 128, NNODES);
    // agg(h) -> Wcat[:, :128]
    k_zero_cols<<<EWB, 256, 0, stream>>>(WC, 256, ND);
    k_agg<<<AGB, 256, 0, stream>>>(M, 128, srcI, dstI, eattr, WC, 256);
    // Wcat[:, 128:] = xm @ l2 + b2
    gemm_wmma<4, 0, false, true><<<GG, 256, S4, stream>>>(
        A, 128, f_l2 + i * 16384, l2b + i * 128, WC + 128, 256, NNODES);
    // w1 = relu(relu(Wcat) @ mlp1 + b)   (K = 256, pre-relu on A load)
    gemm_wmma<8, 1, true, true><<<GG, 256, S8, stream>>>(
        WC, 256, f_m1 + i * 32768, m1b + i * 128, M, 128, NNODES);
    // mask = sigmoid(w1 @ mlp2 + b)      (in-place: A preloaded before stores)
    gemm_wmma<4, 2, false, true><<<GG, 256, S4, stream>>>(
        M, 128, f_m2 + i * 16384, m2b + i * 128, M, 128, NNODES);
    // h2 = (x * mask) @ conv_w -> Wcat[:, 128:]
    k_mul<<<EWB, 256, 0, stream>>>(X, M, A, ND);
    gemm_wmma<4, 0, false, false><<<GG, 256, S4, stream>>>(
        A, 128, f_cw + i * 16384, (const float*)nullptr, WC + 128, 256, NNODES);
    // lin = x @ conv_lin -> A
    gemm_wmma<4, 0, false, false><<<GG, 256, S4, stream>>>(
        X, 128, f_cl + i * 16384, (const float*)nullptr, A, 128, NNODES);
    // agg(h2) -> Wcat[:, :128]
    k_zero_cols<<<EWB, 256, 0, stream>>>(WC, 256, ND);
    k_agg<<<AGB, 256, 0, stream>>>(WC + 128, 256, srcI, dstI, eattr, WC, 256);
    // x = relu((agg + lin) * mask)
    k_combine<<<EWB, 256, 0, stream>>>(WC, A, M, X, ND);
  }

  // post MLP
  gemm_wmma<4, 1, false, true><<<GG, 256, S4, stream>>>(
      X, 128, f_p1, p1b, A, 128, NNODES);
  gemm_wmma<4, 0, false, true><<<GG, 256, S4, stream>>>(
      A, 128, f_p2, p2b, (float*)d_out, 128, NNODES);
}
